// CausalRelationMining_62766652064276
// MI455X (gfx1250) — matrix operand order repaired and checked by reading, not compiled, not run on hardware
//
#include <hip/hip_runtime.h>
#include <hip/hip_bf16.h>

#define DIM   1024
#define HEADS 16
#define HD    64
#define BB    2
#define NN    2048
#define MROWS (BB * NN)   // 4096

typedef __bf16 bf16;
typedef __attribute__((ext_vector_type(16))) __bf16 v16bf;
typedef __attribute__((ext_vector_type(8)))  __bf16 v8bf;
typedef __attribute__((ext_vector_type(8)))  float  v8f;

// ---------- fragment helpers ----------
// A operand (16x32 bf16, M x K): lane = row (l&15); lanes 0-15: K 0-7 & 16-23,
// lanes 16-31: K 8-15 & 24-31.  base points at (row0, k0), ld in elements.
__device__ __forceinline__ v16bf load_fragA(const bf16* __restrict__ base, int ld, int lane) {
  const int r  = lane & 15;
  const int kb = (lane >> 4) << 3;               // 0 or 8
  const bf16* p = base + r * ld + kb;
  v8bf lo = *(const v8bf*)(p);
  v8bf hi = *(const v8bf*)(p + 16);
  return __builtin_shufflevector(lo, hi, 0,1,2,3,4,5,6,7,8,9,10,11,12,13,14,15);
}

// B operand (32x16 bf16, K x N) loaded from a [N,K] (K-contiguous) buffer:
// lane = column (l&15); lanes 0-15: K 0-7 & 16-23, lanes 16-31: K 8-15 & 24-31.
__device__ __forceinline__ v16bf load_fragB(const bf16* __restrict__ base, int ld, int lane) {
  const int n  = lane & 15;
  const int kb = (lane >> 4) << 3;               // 0 or 8
  const bf16* p = base + n * ld + kb;
  v8bf lo = *(const v8bf*)(p);
  v8bf hi = *(const v8bf*)(p + 16);
  return __builtin_shufflevector(lo, hi, 0,1,2,3,4,5,6,7,8,9,10,11,12,13,14,15);
}

__device__ __forceinline__ v8f wmma_bf16(v16bf a, v16bf b, v8f c) {
  return __builtin_amdgcn_wmma_f32_16x16x32_bf16(false, a, false, b, (short)0, c, false, false);
}

// ---------- prep kernels ----------
__global__ __launch_bounds__(256) void prep_act(const float* __restrict__ yf,
                                                const float* __restrict__ yb,
                                                bf16* __restrict__ of,
                                                bf16* __restrict__ ob,
                                                bf16* __restrict__ os) {
  int i = blockIdx.x * blockDim.x + threadIdx.x;
  float a = yf[i], b = yb[i];
  of[i] = (bf16)a;
  ob[i] = (bf16)b;
  os[i] = (bf16)(a + b);
}

__global__ __launch_bounds__(256) void prep_wt(const float* __restrict__ W,
                                               bf16* __restrict__ Wt) {
  int i = blockIdx.x * blockDim.x + threadIdx.x;  // over DIM*DIM
  int n = i / DIM, k = i % DIM;
  Wt[n * DIM + k] = (bf16)W[k * DIM + n];         // Wt[N][K] = W[K][N]
}

// ---------- generic bf16 WMMA GEMM:  out = A[M,K] * Bt[N,K]^T + bias ----------
// mode 0: bf16 out, split-head   [bh][n][64]   (Q, K)
// mode 1: bf16 out, head-transp  [bh][64][n]   (V)
// mode 2: f32  out, row-major    [m][DIM]      (final projection -> d_out)
__global__ __launch_bounds__(128)
void gemm_wmma(const bf16* __restrict__ A, const bf16* __restrict__ Bt,
               const float* __restrict__ bias, void* __restrict__ out, int mode) {
  const int lane = threadIdx.x & 31;
  const int wave = threadIdx.x >> 5;
  const int m0 = blockIdx.y * 128 + (wave >> 1) * 64;
  const int n0 = blockIdx.x * 128 + (wave & 1) * 64;

  v8f acc[4][4] = {};
  for (int k0 = 0; k0 < DIM; k0 += 32) {
    v16bf af[4], bfr[4];
#pragma unroll
    for (int mf = 0; mf < 4; ++mf)
      af[mf] = load_fragA(A + (m0 + mf * 16) * DIM + k0, DIM, lane);
#pragma unroll
    for (int nf = 0; nf < 4; ++nf)
      bfr[nf] = load_fragB(Bt + (n0 + nf * 16) * DIM + k0, DIM, lane);
#pragma unroll
    for (int mf = 0; mf < 4; ++mf)
#pragma unroll
      for (int nf = 0; nf < 4; ++nf)
        acc[mf][nf] = wmma_bf16(af[mf], bfr[nf], acc[mf][nf]);
  }

  const int hi = (lane >> 4) << 3;   // row offset 0 / 8
  const int nn = lane & 15;
#pragma unroll
  for (int mf = 0; mf < 4; ++mf) {
#pragma unroll
    for (int nf = 0; nf < 4; ++nf) {
      const int ncol = n0 + nf * 16 + nn;
      const float bv = bias[ncol];
#pragma unroll
      for (int r = 0; r < 8; ++r) {
        const int mrow = m0 + mf * 16 + hi + r;
        const float val = acc[mf][nf][r] + bv;
        if (mode == 2) {
          ((float*)out)[mrow * DIM + ncol] = val;
        } else {
          const int b  = mrow >> 11;          // / NN
          const int i  = mrow & (NN - 1);
          const int hh = ncol >> 6;
          const int dd = ncol & 63;
          const int bh = b * HEADS + hh;
          if (mode == 0)
            ((bf16*)out)[(bh * NN + i) * HD + dd] = (bf16)val;
          else
            ((bf16*)out)[(bh * HD + dd) * NN + i] = (bf16)val;
        }
      }
    }
  }
}

// ---------- flash attention (streaming softmax, all WMMA) ----------
__global__ __launch_bounds__(128)
void attn_wmma(const bf16* __restrict__ Qh, const bf16* __restrict__ Kh,
               const bf16* __restrict__ Vt,
               const float* __restrict__ yf, const float* __restrict__ yb,
               bf16* __restrict__ attout) {
  __shared__ bf16 lds_p[4][32 * 64];             // per-wave P tile (32x64)
  const int lane = threadIdx.x & 31;
  const int wave = threadIdx.x >> 5;
  const int bh   = blockIdx.x >> 4;              // 0..31
  const int it   = blockIdx.x & 15;
  const int row0 = it * 128 + wave * 32;
  const int b = bh >> 4, h = bh & 15;

  const bf16* Qb = Qh + (bh * NN + row0) * HD;
  const bf16* Kb = Kh + (size_t)bh * NN * HD;
  const bf16* Vb = Vt + (size_t)bh * HD * NN;

  // Q fragments are reused for the whole j loop
  v16bf qf[2][2];
#pragma unroll
  for (int mf = 0; mf < 2; ++mf)
#pragma unroll
    for (int kf = 0; kf < 2; ++kf)
      qf[mf][kf] = load_fragA(Qb + (mf * 16) * HD + kf * 32, HD, lane);

  v8f o[2][4] = {};
  float mrow[2][8], srow[2][8];
#pragma unroll
  for (int mf = 0; mf < 2; ++mf)
#pragma unroll
    for (int r = 0; r < 8; ++r) { mrow[mf][r] = -1e30f; srow[mf][r] = 0.f; }

  const float scale = 0.125f;                    // 1/sqrt(64)
  const int hi = (lane >> 4) << 3;
  const int nn = lane & 15;
  bf16* lp = lds_p[wave];

  for (int j0 = 0; j0 < NN; j0 += 64) {
    // ---- S = Q K^T (scaled) ----
    v16bf kfag[4][2];
#pragma unroll
    for (int nf = 0; nf < 4; ++nf)
#pragma unroll
      for (int kf = 0; kf < 2; ++kf)
        kfag[nf][kf] = load_fragB(Kb + (j0 + nf * 16) * HD + kf * 32, HD, lane);

    v8f s[2][4] = {};
#pragma unroll
    for (int mf = 0; mf < 2; ++mf)
#pragma unroll
      for (int nf = 0; nf < 4; ++nf)
#pragma unroll
        for (int kf = 0; kf < 2; ++kf)
          s[mf][nf] = wmma_bf16(qf[mf][kf], kfag[nf][kf], s[mf][nf]);

#pragma unroll
    for (int mf = 0; mf < 2; ++mf)
#pragma unroll
      for (int nf = 0; nf < 4; ++nf)
#pragma unroll
        for (int r = 0; r < 8; ++r)
          s[mf][nf][r] *= scale;

    // ---- streaming softmax: row max / exp / row sum ----
    float alpha[2][8];
#pragma unroll
    for (int mf = 0; mf < 2; ++mf) {
#pragma unroll
      for (int r = 0; r < 8; ++r) {
        float mx = s[mf][0][r];
#pragma unroll
        for (int nf = 1; nf < 4; ++nf) mx = fmaxf(mx, s[mf][nf][r]);
        mx = fmaxf(mx, __shfl_xor(mx, 1, 32));
        mx = fmaxf(mx, __shfl_xor(mx, 2, 32));
        mx = fmaxf(mx, __shfl_xor(mx, 4, 32));
        mx = fmaxf(mx, __shfl_xor(mx, 8, 32));
        const float newm = fmaxf(mrow[mf][r], mx);
        const float al = __expf(mrow[mf][r] - newm);
        mrow[mf][r] = newm;
        alpha[mf][r] = al;
        float ps = 0.f;
#pragma unroll
        for (int nf = 0; nf < 4; ++nf) {
          float p = __expf(s[mf][nf][r] - newm);
          s[mf][nf][r] = p;
          ps += p;
        }
        ps += __shfl_xor(ps, 1, 32);
        ps += __shfl_xor(ps, 2, 32);
        ps += __shfl_xor(ps, 4, 32);
        ps += __shfl_xor(ps, 8, 32);
        srow[mf][r] = srow[mf][r] * al + ps;
      }
    }

    // ---- P: C-layout -> LDS row-major -> A-layout fragments ----
#pragma unroll
    for (int mf = 0; mf < 2; ++mf)
#pragma unroll
      for (int nf = 0; nf < 4; ++nf)
#pragma unroll
        for (int r = 0; r < 8; ++r)
          lp[(mf * 16 + hi + r) * 64 + nf * 16 + nn] = (bf16)s[mf][nf][r];
    asm volatile("s_wait_dscnt 0" ::: "memory");   // wave-local LDS RAW

    v16bf pf[2][2];
#pragma unroll
    for (int mf = 0; mf < 2; ++mf)
#pragma unroll
      for (int kf = 0; kf < 2; ++kf)
        pf[mf][kf] = load_fragA(lp + (mf * 16) * 64 + kf * 32, 64, lane);

    // ---- O = O*alpha + P @ V ----
#pragma unroll
    for (int mf = 0; mf < 2; ++mf)
#pragma unroll
      for (int nf = 0; nf < 4; ++nf)
#pragma unroll
        for (int r = 0; r < 8; ++r)
          o[mf][nf][r] *= alpha[mf][r];

    v16bf vfg[4][2];
#pragma unroll
    for (int nf = 0; nf < 4; ++nf)
#pragma unroll
      for (int kf = 0; kf < 2; ++kf)
        vfg[nf][kf] = load_fragB(Vb + (nf * 16) * NN + j0 + kf * 32, NN, lane);

#pragma unroll
    for (int mf = 0; mf < 2; ++mf)
#pragma unroll
      for (int nf = 0; nf < 4; ++nf)
#pragma unroll
        for (int kf = 0; kf < 2; ++kf)
          o[mf][nf] = wmma_bf16(pf[mf][kf], vfg[nf][kf], o[mf][nf]);
  }

  // ---- finalize: O/rowsum + residual -> bf16 for the output GEMM ----
#pragma unroll
  for (int mf = 0; mf < 2; ++mf) {
#pragma unroll
    for (int r = 0; r < 8; ++r) {
      const float inv = 1.f / srow[mf][r];
      const int i = row0 + mf * 16 + hi + r;
#pragma unroll
      for (int nf = 0; nf < 4; ++nf) {
        const int col = h * 64 + nf * 16 + nn;
        const int idx = (b * NN + i) * DIM + col;
        const float val = o[mf][nf][r] * inv + yf[idx] + yb[idx];
        attout[idx] = (bf16)val;
      }
    }
  }
}

// ---------- host ----------
extern "C" void kernel_launch(void* const* d_in, const int* in_sizes, int n_in,
                              void* d_out, int out_size, void* d_ws, size_t ws_size,
                              hipStream_t stream) {
  (void)in_sizes; (void)n_in; (void)out_size; (void)ws_size;
  const float* yf = (const float*)d_in[0];
  const float* yb = (const float*)d_in[1];
  const float* Wq = (const float*)d_in[2];
  const float* bq = (const float*)d_in[3];
  const float* Wk = (const float*)d_in[4];
  const float* bk = (const float*)d_in[5];
  const float* Wv = (const float*)d_in[6];
  const float* bv = (const float*)d_in[7];
  const float* Wo = (const float*)d_in[8];
  const float* bo = (const float*)d_in[9];

  char* ws = (char*)d_ws;
  size_t off = 0;
  auto carve = [&](size_t bytes) -> void* {
    void* p = ws + off;
    off += (bytes + 255) & ~(size_t)255;
    return p;
  };
  const size_t actB = (size_t)MROWS * DIM * sizeof(bf16);   // 8 MiB each
  const size_t wB   = (size_t)DIM * DIM * sizeof(bf16);     // 2 MiB each
  bf16* yf_b = (bf16*)carve(actB);
  bf16* yb_b = (bf16*)carve(actB);
  bf16* ys_b = (bf16*)carve(actB);
  bf16* WqT  = (bf16*)carve(wB);
  bf16* WkT  = (bf16*)carve(wB);
  bf16* WvT  = (bf16*)carve(wB);
  bf16* WoT  = (bf16*)carve(wB);
  bf16* Qh   = (bf16*)carve(actB);
  bf16* Kh   = (bf16*)carve(actB);
  bf16* Vt   = (bf16*)carve(actB);
  bf16* att  = (bf16*)carve(actB);

  prep_act<<<(MROWS * DIM) / 256, 256, 0, stream>>>(yf, yb, yf_b, yb_b, ys_b);
  prep_wt<<<(DIM * DIM) / 256, 256, 0, stream>>>(Wq, WqT);
  prep_wt<<<(DIM * DIM) / 256, 256, 0, stream>>>(Wk, WkT);
  prep_wt<<<(DIM * DIM) / 256, 256, 0, stream>>>(Wv, WvT);
  prep_wt<<<(DIM * DIM) / 256, 256, 0, stream>>>(Wo, WoT);

  dim3 gg(DIM / 128, MROWS / 128);
  gemm_wmma<<<gg, 128, 0, stream>>>(yf_b, WqT, bq, Qh, 0);
  gemm_wmma<<<gg, 128, 0, stream>>>(yb_b, WkT, bk, Kh, 0);
  gemm_wmma<<<gg, 128, 0, stream>>>(ys_b, WvT, bv, Vt, 1);

  attn_wmma<<<BB * HEADS * (NN / 128), 128, 0, stream>>>(Qh, Kh, Vt, yf, yb, att);

  gemm_wmma<<<gg, 128, 0, stream>>>(att, WoT, bo, (float*)d_out, 2);
}
